// SparseSelfAttention_17660905521473
// MI455X (gfx1250) — matrix operand order repaired and checked by reading, not compiled, not run on hardware
//
#include <hip/hip_runtime.h>
#include <hip/hip_bf16.h>

// Problem constants (match reference)
#define BB 2
#define HH 16
#define SS 2048
#define DD 64
#define NB 128   // S / BLOCK

typedef __attribute__((ext_vector_type(16))) __bf16 v16bf;
typedef __attribute__((ext_vector_type(8)))  float  v8f;
typedef __attribute__((ext_vector_type(8)))  int    v8i;

// pack two f32 -> bf16x2 by truncation: one v_perm_b32
// result = { hi16(b) , hi16(a) }  (a -> low half, b -> high half)
__device__ __forceinline__ int pkbf(float a, float b) {
#if __has_builtin(__builtin_amdgcn_perm)
  return (int)__builtin_amdgcn_perm(__float_as_uint(b), __float_as_uint(a),
                                    0x07060302u);
#else
  return (int)((__float_as_uint(a) >> 16) | (__float_as_uint(b) & 0xffff0000u));
#endif
}

// S^T(16x16) = K_tile(16x64) x Q^T(64x16) for one key block.
// A 16x32 bf16 layout: row = lane&15, VGPR r holds K=(r<4?2r:2r+8)+8*half
// -> lane reads two contiguous 8-float runs per 32-d chunk => float4 loads.
__device__ __forceinline__ v8f score_tile(const float* __restrict__ Kp,
                                          const v8i* qreg, int hh) {
  v8f c = (v8f){};
#pragma unroll
  for (int cc = 0; cc < 2; ++cc) {
    const float* p = Kp + cc * 32 + 8 * hh;
    const float4 u0 = *reinterpret_cast<const float4*>(p);
    const float4 u1 = *reinterpret_cast<const float4*>(p + 4);
    const float4 u2 = *reinterpret_cast<const float4*>(p + 16);
    const float4 u3 = *reinterpret_cast<const float4*>(p + 20);
    v8i a;
    a[0] = pkbf(u0.x, u0.y); a[1] = pkbf(u0.z, u0.w);
    a[2] = pkbf(u1.x, u1.y); a[3] = pkbf(u1.z, u1.w);
    a[4] = pkbf(u2.x, u2.y); a[5] = pkbf(u2.z, u2.w);
    a[6] = pkbf(u3.x, u3.y); a[7] = pkbf(u3.z, u3.w);
    c = __builtin_amdgcn_wmma_f32_16x16x32_bf16(
        false, __builtin_bit_cast(v16bf, a), false,
        __builtin_bit_cast(v16bf, qreg[cc]), (short)0, c, false, false);
  }
  return c;
}

// One wave (32 lanes) per (batch*head, query-block).
// Computes S^T = K*Q^T and O^T = V^T*P^T so softmax stats live per-lane.
__global__ __launch_bounds__(32)
void bs_attn_kernel(const float* __restrict__ Q, const float* __restrict__ Km,
                    const float* __restrict__ Vm, const int* __restrict__ lay,
                    float* __restrict__ O) {
  const int qb   = blockIdx.x;        // query block 0..NB-1
  const int bh   = blockIdx.y;        // 0..B*H-1
  const int h    = bh & (HH - 1);
  const int lane = threadIdx.x;
  const int hh   = lane >> 4;         // wave half
  const int ln   = lane & 15;

  // ---- build active key-block list (wave-uniform) ----
  __shared__ int s_act[NB + 1];
  if (lane == 0) {
    const int* lrow = lay + ((size_t)h * NB + qb) * NB;
    int na = 0;
    for (int j = 0; j < NB; ++j) if (lrow[j] != 0) s_act[na++] = j;
    s_act[NB] = na;
  }
  __syncthreads();
  const int na = s_act[NB];

  const size_t bhBase = (size_t)bh * SS * DD;
  const size_t qrow   = bhBase + (size_t)(qb * 16 + ln) * DD; // this lane's query row

  // ---- Q^T as B-operand, two d-chunks; lane reads two 8-float runs ----
  // B 32x16 bf16 layout: col = lane&15, VGPR r holds K=(r<4?2r:2r+8)+8*half, +1
  v8i qreg[2];
#pragma unroll
  for (int c = 0; c < 2; ++c) {
    const float* p = Q + qrow + c * 32 + 8 * hh;
    const float4 u0 = *reinterpret_cast<const float4*>(p);
    const float4 u1 = *reinterpret_cast<const float4*>(p + 4);
    const float4 u2 = *reinterpret_cast<const float4*>(p + 16);
    const float4 u3 = *reinterpret_cast<const float4*>(p + 20);
    qreg[c][0] = pkbf(u0.x, u0.y); qreg[c][1] = pkbf(u0.z, u0.w);
    qreg[c][2] = pkbf(u1.x, u1.y); qreg[c][3] = pkbf(u1.z, u1.w);
    qreg[c][4] = pkbf(u2.x, u2.y); qreg[c][5] = pkbf(u2.z, u2.w);
    qreg[c][6] = pkbf(u3.x, u3.y); qreg[c][7] = pkbf(u3.z, u3.w);
  }

  v8f o[4];                       // O^T accumulators: 4 M-tiles of 16 d-rows
#pragma unroll
  for (int mt = 0; mt < 4; ++mt) o[mt] = (v8f){};
  float Mrow = -__builtin_inff();
  float Lrow = 0.0f;
  const float kscale = 0.125f * 1.44269504088896f; // 1/sqrt(D) * log2(e)

  for (int t = 0; t < na; t += 2) {
    const int  j0   = s_act[t];
    const bool has1 = (t + 1) < na;
    const int  j1   = has1 ? s_act[t + 1] : j0;

    // prefetch next pair's K/V rows into cache hierarchy
    if (t + 2 < na) {
      const size_t nOff = bhBase + (size_t)(s_act[t + 2] * 16 + ln) * DD;
      __builtin_prefetch((const char*)(Km + nOff), 0, 3);
      __builtin_prefetch((const char*)(Vm + nOff), 0, 3);
    }

    // ---- scores: S^T = K_tile x Q^T (C layout: row=key kk, col=query q=ln) ----
    v8f c0 = score_tile(Km + bhBase + (size_t)(j0 * 16 + ln) * DD, qreg, hh);
    v8f c1 = (v8f){};
    if (has1)
      c1 = score_tile(Km + bhBase + (size_t)(j1 * 16 + ln) * DD, qreg, hh);

    // ---- online softmax over the (up to) 32 keys; q = lane&15 ----
    // Missing second block => scores -inf => probabilities exactly 0.
    const float bias1 = has1 ? 0.0f : -__builtin_inff();
    float f0[8], f1[8];
#pragma unroll
    for (int j = 0; j < 8; ++j) {
      f0[j] = c0[j];
      f1[j] = has1 ? c1[j] : bias1;
    }
    float mloc = -__builtin_inff();
#pragma unroll
    for (int j = 0; j < 8; ++j) mloc = fmaxf(mloc, fmaxf(f0[j], f1[j]));
    mloc = fmaxf(mloc, __shfl_xor(mloc, 16, 32));   // other 8 keys of each block
    const float Mn    = fmaxf(Mrow, mloc * kscale);
    const float alpha = __builtin_amdgcn_exp2f(Mrow - Mn);
    float p0[8], p1[8], rs = 0.0f;
#pragma unroll
    for (int j = 0; j < 8; ++j) {
      p0[j] = __builtin_amdgcn_exp2f(f0[j] * kscale - Mn);
      p1[j] = __builtin_amdgcn_exp2f(f1[j] * kscale - Mn);  // 0 when masked
      rs += p0[j] + p1[j];
    }
    rs += __shfl_xor(rs, 16, 32);
    Lrow = Lrow * alpha + rs;
    Mrow = Mn;
#pragma unroll
    for (int mt = 0; mt < 4; ++mt) o[mt] = o[mt] * alpha;  // v_pk_mul_f32

    // ---- repack P^T: pack to bf16x2 first, then half-swap packed words ----
    int w0[4], w1[4], w0s[4], w1s[4];
#pragma unroll
    for (int r = 0; r < 4; ++r) {
      w0[r] = pkbf(p0[2 * r], p0[2 * r + 1]);
      w1[r] = pkbf(p1[2 * r], p1[2 * r + 1]);
    }
#pragma unroll
    for (int r = 0; r < 4; ++r) {
      w0s[r] = __shfl_xor(w0[r], 16, 32);
      w1s[r] = __shfl_xor(w1[r], 16, 32);
    }
    v8i bp;
#pragma unroll
    for (int r = 0; r < 4; ++r) {     // B rows K=2r+8*hh,+1 of j0
      bp[r]     = w0[r];
      bp[r + 4] = w1[r];   // B rows K=16+2r+8*hh,+1 (block j1)
    }
    const v16bf Bp = __builtin_bit_cast(v16bf, bp);

    // ---- O^T += V^T x P^T : 4 M-tiles (d in chunks of 16), K=32 ----
    // A reg r<4 reads rows kk=2r+8*hh of block j0; reg r>=4 reads rows
    // kk-16=2(r-4)+8*hh of block j1; column = mt*16 + ln.  Everything is a
    // compile-time offset from two lane base pointers -> pure imm-offset loads.
    const float* pb0 = Vm + bhBase + (size_t)(j0 * 16 + 8 * hh) * DD + ln;
    const float* pb1 = Vm + bhBase + (size_t)(j1 * 16 + 8 * hh) * DD + ln;
#pragma unroll
    for (int mt = 0; mt < 4; ++mt) {
      v8i a;
#pragma unroll
      for (int r = 0; r < 4; ++r) {
        const int off = 2 * r * DD + mt * 16;
        a[r]     = pkbf(pb0[off], pb0[off + DD]);  // rows kk, kk+1 of j0
        a[r + 4] = pkbf(pb1[off], pb1[off + DD]);  // rows kk, kk+1 of j1
      }
      o[mt] = __builtin_amdgcn_wmma_f32_16x16x32_bf16(
          false, __builtin_bit_cast(v16bf, a), false, Bp, (short)0, o[mt],
          false, false);
    }
  }

  // ---- normalize and store O (row q = ln); d runs contiguously per lane ----
  const float inv = 1.0f / Lrow;
#pragma unroll
  for (int mt = 0; mt < 4; ++mt) {
    const v8f on = o[mt] * inv;                    // v_pk_mul_f32
    float4 s0, s1;
    s0.x = on[0]; s0.y = on[1]; s0.z = on[2]; s0.w = on[3];
    s1.x = on[4]; s1.y = on[5]; s1.z = on[6]; s1.w = on[7];
    float* dst = O + qrow + mt * 16 + 8 * hh;      // d = mt*16 + 8*hh + j
    *reinterpret_cast<float4*>(dst)     = s0;
    *reinterpret_cast<float4*>(dst + 4) = s1;
  }
}

extern "C" void kernel_launch(void* const* d_in, const int* in_sizes, int n_in,
                              void* d_out, int out_size, void* d_ws, size_t ws_size,
                              hipStream_t stream) {
  const float* Q  = (const float*)d_in[0];
  const float* K  = (const float*)d_in[1];
  const float* V  = (const float*)d_in[2];
  const int* lay  = (const int*)d_in[3];
  float* out      = (float*)d_out;
  (void)in_sizes; (void)n_in; (void)out_size; (void)d_ws; (void)ws_size;

  dim3 grid(NB, BB * HH);   // 128 query-blocks x 32 (batch*head)
  bs_attn_kernel<<<grid, 32, 0, stream>>>(Q, K, V, lay, out);
}
